// ProbAttention_62603443306806
// MI455X (gfx1250) — matrix operand order repaired and checked by reading, not compiled, not run on hardware
//
#include <hip/hip_runtime.h>
#include <hip/hip_bf16.h>

typedef __attribute__((ext_vector_type(16))) _Float16 v16h;
typedef __attribute__((ext_vector_type(8)))  _Float16 v8h;
typedef __attribute__((ext_vector_type(8)))  float    v8f;
typedef __attribute__((ext_vector_type(4)))  int      v4i;

#define B_  8
#define L_  4096
#define D_  512
#define H_  8
#define DK_ 64
#define S_  512
#define E_  512   // H*DK

// ---------------------------------------------------------------- WMMA core
__device__ __forceinline__ v8f wmma16(v16h a, v16h b, v8f c) {
  // D = A(16x32 f16) * B(32x16 f16) + C(16x16 f32)
  return __builtin_amdgcn_wmma_f32_16x16x32_f16(false, a, false, b, (short)0, c,
                                                false, false);
}

// A-fragment (16x32 f16): lane L: M = L%16, khalf = L/16.
// element e -> K = kk + khalf*16 + e
__device__ __forceinline__ v16h frag_a_f16(const _Float16* __restrict__ row,
                                           int kk, int khalf) {
  v16h a;
#pragma unroll
  for (int e = 0; e < 8; ++e) {
    a[e]     = row[kk + khalf * 16 + e];
    a[e + 8] = row[kk + khalf * 16 + 8 + e];
  }
  return a;
}
// B-fragment (32x16 f16) from a [N][K] row: element e -> k = kk + khalf*16 + e
__device__ __forceinline__ v16h frag_b(const _Float16* __restrict__ rowN, int kk,
                                       int khalf) {
  return *(const v16h*)(rowN + kk + khalf * 16);
}

// ------------------------ gfx1250 async global->LDS copy (ASYNCcnt tracked)
typedef __attribute__((address_space(1))) v4i gv4i;
typedef __attribute__((address_space(3))) v4i lv4i;

__device__ __forceinline__ void async_copy_b128(const _Float16* gsrc,
                                                _Float16* ldst) {
#if __has_builtin(__builtin_amdgcn_global_load_async_to_lds_b128)
  __builtin_amdgcn_global_load_async_to_lds_b128((gv4i*)gsrc, (lv4i*)ldst, 0,
                                                 0);
#else
  *(v8h*)ldst = *(const v8h*)gsrc;  // sync fallback (16B)
#endif
}
__device__ __forceinline__ void async_wait0() {
#if __has_builtin(__builtin_amdgcn_s_wait_asynccnt)
  __builtin_amdgcn_s_wait_asynccnt(0);
#else
  asm volatile("s_wait_asynccnt 0" ::: "memory");
#endif
}

// Stage a 64(N) x 128(K) f16 B tile into LDS: 1024 b128 transfers / 256 thr.
__device__ __forceinline__ void stage_b(const _Float16* __restrict__ WT, int K,
                                        int n0, int kc, _Float16 (*Bs)[128]) {
#pragma unroll
  for (int it = 0; it < 4; ++it) {
    int idx = it * 256 + threadIdx.x;  // 0..1023
    int row = idx >> 4;                // 0..63
    int col8 = (idx & 15) << 3;        // 0..120 step 8
    async_copy_b128(WT + (size_t)(n0 + row) * K + kc + col8, &Bs[row][col8]);
  }
}

// ------------------------------------------------------- prep: W -> WT (f16)
__global__ void __launch_bounds__(256) k_wt(const float* __restrict__ W,
                                            _Float16* __restrict__ WT, int K,
                                            int N) {
  int idx = blockIdx.x * 256 + threadIdx.x;
  if (idx < K * N) {
    int k = idx / N, n = idx % N;
    WT[(size_t)n * K + k] = (_Float16)W[idx];
  }
}

// x (f32) -> xh (f16), once
__global__ void __launch_bounds__(256) k_xh(const float* __restrict__ X,
                                            _Float16* __restrict__ Xh, int n) {
  int i = blockIdx.x * 256 + threadIdx.x;
  if (i < n) Xh[i] = (_Float16)X[i];
}

// idx2[j] = sample_indices[ clip(sample_indices[j], 0, S-1) ]  (gather fusion)
__global__ void __launch_bounds__(256) k_idx2(const int* __restrict__ sidx,
                                              int* __restrict__ idx2) {
  int j = blockIdx.x * 256 + threadIdx.x;
  if (j < S_) {
    int g = sidx[j];
    g = g < 0 ? 0 : (g > S_ - 1 ? S_ - 1 : g);
    int v = sidx[g];
    v = v < 0 ? 0 : (v > L_ - 1 ? L_ - 1 : v);
    idx2[j] = v;
  }
}

// ------------------- GEMM (f16 A x f16 B -> f16 C), B staged via async LDS
__global__ void __launch_bounds__(256)
k_gemm_hw(const _Float16* __restrict__ A, const _Float16* __restrict__ WT,
          _Float16* __restrict__ Cout, int K, int N) {
  __shared__ _Float16 Bs[64][128];
  const int lane = threadIdx.x & 31, wave = threadIdx.x >> 5;
  const int lane16 = lane & 15, khalf = lane >> 4;
  const int m0 = blockIdx.x * 128 + wave * 16;
  const int n0 = blockIdx.y * 64;
  const _Float16* arow = A + (size_t)(m0 + lane16) * K;
  v8f acc[4] = {};
  for (int kc = 0; kc < K; kc += 128) {
    __syncthreads();                 // previous chunk consumed
    stage_b(WT, K, n0, kc, Bs);      // async global -> LDS
    async_wait0();
    __syncthreads();                 // tile visible to all 8 waves
#pragma unroll
    for (int kkl = 0; kkl < 128; kkl += 32) {
      v16h a = frag_a_f16(arow, kc + kkl, khalf);
#pragma unroll
      for (int nt = 0; nt < 4; ++nt)
        acc[nt] =
            wmma16(a, *(const v16h*)&Bs[nt * 16 + lane16][kkl + khalf * 16],
                   acc[nt]);
    }
  }
  const int rb = (lane & 16) ? 8 : 0;
#pragma unroll
  for (int i = 0; i < 8; ++i)
#pragma unroll
    for (int nt = 0; nt < 4; ++nt)
      Cout[(size_t)(m0 + rb + i) * N + n0 + nt * 16 + lane16] =
          (_Float16)acc[nt][i];
}

// ------------------------------------------- gather Ksg[j]=K[idx2[j]] (row-major)
// and Vsht[d][j]=V[idx[j]][d] (transposed so PV B-fragments are contiguous)
__global__ void __launch_bounds__(256)
k_gather(const _Float16* __restrict__ Kh, const _Float16* __restrict__ Vh,
         const int* __restrict__ sidx, const int* __restrict__ idx2,
         _Float16* __restrict__ Ksg, _Float16* __restrict__ Vsht) {
  int t = blockIdx.x * 256 + threadIdx.x;
  if (t >= B_ * H_ * S_ * DK_) return;
  int d = t & 63, j = (t >> 6) & (S_ - 1), bh = t >> 15;
  int b = bh >> 3, h = bh & 7;
  int lk = idx2[j];
  int lv = sidx[j];
  lv = lv < 0 ? 0 : (lv > L_ - 1 ? L_ - 1 : lv);
  Ksg[((size_t)bh * S_ + j) * DK_ + d] =
      Kh[((size_t)b * L_ + lk) * E_ + h * DK_ + d];
  Vsht[((size_t)bh * DK_ + d) * S_ + j] =
      Vh[((size_t)b * L_ + lv) * E_ + h * DK_ + d];
}

// ----------------------- fused attention: scores->softmax->PV, flash-style
__global__ void __launch_bounds__(256)
k_attn(const _Float16* __restrict__ Qh, const _Float16* __restrict__ Ksg,
       const _Float16* __restrict__ Vsht, _Float16* __restrict__ Oh) {
  __shared__ _Float16 Pbuf[8][16][64];
  const int lane = threadIdx.x & 31, wave = threadIdx.x >> 5;
  const int lane16 = lane & 15, khalf = lane >> 4;
  const int wid = blockIdx.x * 8 + wave;  // 16384 wave tasks
  const int bh = wid >> 8;                // 256 row tiles per (b,h)
  const int m0 = (wid & 255) * 16;
  const int b = bh >> 3, h = bh & 7;

  const _Float16* qrow = Qh + ((size_t)b * L_ + m0 + lane16) * E_ + h * DK_;
  const v16h qa0 = frag_a_f16(qrow, 0, khalf);
  const v16h qa1 = frag_a_f16(qrow, 32, khalf);
  const _Float16* ksgB = Ksg + (size_t)bh * S_ * DK_;
  const _Float16* vsB  = Vsht + (size_t)bh * DK_ * S_;

  v8f acc[4] = {};
  float mrow[8], lrow[8];
#pragma unroll
  for (int i = 0; i < 8; ++i) { mrow[i] = -1e30f; lrow[i] = 0.f; }
  const int rb = (lane & 16) ? 8 : 0;
  const float scale = 0.125f;  // 1/sqrt(64)

  for (int j0 = 0; j0 < S_; j0 += 64) {
    // scores chunk [16 x 64] = Q(16x64) x Ksg_chunk^T
    v8f s[4] = {};
#pragma unroll
    for (int nt = 0; nt < 4; ++nt) {
      const _Float16* kn = ksgB + (size_t)(j0 + nt * 16 + lane16) * DK_;
      s[nt] = wmma16(qa0, frag_b(kn, 0, khalf), s[nt]);
      s[nt] = wmma16(qa1, frag_b(kn, 32, khalf), s[nt]);
    }
    // online softmax (rows live in VGPR i across the 16-lane group)
#pragma unroll
    for (int i = 0; i < 8; ++i) {
      float cm = -1e30f;
#pragma unroll
      for (int nt = 0; nt < 4; ++nt) {
        s[nt][i] *= scale;
        cm = fmaxf(cm, s[nt][i]);
      }
#pragma unroll
      for (int off = 1; off < 16; off <<= 1)
        cm = fmaxf(cm, __shfl_xor(cm, off, 32));
      float mnew = fmaxf(mrow[i], cm);
      float corr = __expf(mrow[i] - mnew);
      mrow[i] = mnew;
      float psum = 0.f;
#pragma unroll
      for (int nt = 0; nt < 4; ++nt) {
        float p = __expf(s[nt][i] - mnew);
        s[nt][i] = p;
        psum += p;
      }
#pragma unroll
      for (int off = 1; off < 16; off <<= 1) psum += __shfl_xor(psum, off, 32);
      lrow[i] = lrow[i] * corr + psum;
#pragma unroll
      for (int nt = 0; nt < 4; ++nt) acc[nt][i] *= corr;
    }
    // P tile: C-layout -> LDS -> A-layout (same wave; DS ops are in-order)
#pragma unroll
    for (int i = 0; i < 8; ++i)
#pragma unroll
      for (int nt = 0; nt < 4; ++nt)
        Pbuf[wave][rb + i][nt * 16 + lane16] = (_Float16)s[nt][i];
    asm volatile("s_wait_dscnt 0" ::: "memory");
    // acc += P(16x64) x Vs_chunk(64x64)
#pragma unroll
    for (int ks = 0; ks < 2; ++ks) {
      v16h pa = frag_a_f16(&Pbuf[wave][lane16][0], ks * 32, khalf);
#pragma unroll
      for (int dt = 0; dt < 4; ++dt) {
        const _Float16* vn = vsB + (size_t)(dt * 16 + lane16) * S_ + j0;
        acc[dt] = wmma16(pa, frag_b(vn, ks * 32, khalf), acc[dt]);
      }
    }
  }
  // normalize + store f16 to [B, L, E]
#pragma unroll
  for (int i = 0; i < 8; ++i) {
    float inv = 1.0f / lrow[i];
#pragma unroll
    for (int dt = 0; dt < 4; ++dt)
      Oh[((size_t)b * L_ + m0 + rb + i) * E_ + h * DK_ + dt * 16 + lane16] =
          (_Float16)(acc[dt][i] * inv);
  }
}

// --------------------- conv1d(k=3, SAME) as GEMM K=1536, fused bias + ELU
__global__ void __launch_bounds__(256)
k_conv(const _Float16* __restrict__ AO, const _Float16* __restrict__ WT,
       const float* __restrict__ bias, _Float16* __restrict__ Yc) {
  __shared__ _Float16 Bs[64][128];
  const int lane = threadIdx.x & 31, wave = threadIdx.x >> 5;
  const int lane16 = lane & 15, khalf = lane >> 4;
  const int m0 = blockIdx.x * 128 + wave * 16;
  const int n0 = blockIdx.y * 64;
  const int m = m0 + lane16;
  const int b = m >> 12;  // / 4096
  const int t = m & (L_ - 1);
  v8f acc[4] = {};
  for (int kc = 0; kc < 3 * E_; kc += 128) {
    __syncthreads();
    stage_b(WT, 3 * E_, n0, kc, Bs);
    async_wait0();
    __syncthreads();
#pragma unroll
    for (int kkl = 0; kkl < 128; kkl += 32) {
      const int kk = kc + kkl;
      v16h a;
      {
        int kr = kk + khalf * 16;
        int slab = kr >> 9, ci = kr & 511, tt = t + slab - 1;
        bool ok = (unsigned)tt < (unsigned)L_;
        const _Float16* src = AO + ((size_t)b * L_ + (ok ? tt : 0)) * E_ + ci;
#pragma unroll
        for (int e = 0; e < 8; ++e) a[e] = ok ? src[e] : (_Float16)0.0f;
      }
      {
        int kr = kk + 8 + khalf * 16;
        int slab = kr >> 9, ci = kr & 511, tt = t + slab - 1;
        bool ok = (unsigned)tt < (unsigned)L_;
        const _Float16* src = AO + ((size_t)b * L_ + (ok ? tt : 0)) * E_ + ci;
#pragma unroll
        for (int e = 0; e < 8; ++e) a[e + 8] = ok ? src[e] : (_Float16)0.0f;
      }
#pragma unroll
      for (int nt = 0; nt < 4; ++nt)
        acc[nt] =
            wmma16(a, *(const v16h*)&Bs[nt * 16 + lane16][kkl + khalf * 16],
                   acc[nt]);
    }
  }
  const int rb = (lane & 16) ? 8 : 0;
#pragma unroll
  for (int i = 0; i < 8; ++i)
#pragma unroll
    for (int nt = 0; nt < 4; ++nt) {
      int n = n0 + nt * 16 + lane16;
      float y = acc[nt][i] + bias[n];
      y = (y > 0.f) ? y : (__expf(y) - 1.f);
      Yc[(size_t)(m0 + rb + i) * E_ + n] = (_Float16)y;
    }
}

// ------------------------------------------------------------- maxpool by 2
__global__ void __launch_bounds__(256) k_pool(const _Float16* __restrict__ Yc,
                                              _Float16* __restrict__ P) {
  int t = blockIdx.x * 256 + threadIdx.x;  // over B*(L/2)*E = 8388608
  int c = t & 511;
  int u = t >> 9;  // B*(L/2)
  int b = u >> 11, uu = u & 2047;
  const _Float16* p = Yc + ((size_t)b * L_ + uu * 2) * E_ + c;
  P[t] = (_Float16)fmaxf((float)p[0], (float)p[E_]);
}

// ---------------- final dense (async-staged B): f16 x f16 + bias -> f32 out
__global__ void __launch_bounds__(256)
k_final(const _Float16* __restrict__ A, const _Float16* __restrict__ WT,
        const float* __restrict__ bias, float* __restrict__ Out) {
  __shared__ _Float16 Bs[64][128];
  const int lane = threadIdx.x & 31, wave = threadIdx.x >> 5;
  const int lane16 = lane & 15, khalf = lane >> 4;
  const int m0 = blockIdx.x * 128 + wave * 16;
  const int n0 = blockIdx.y * 64;
  const _Float16* arow = A + (size_t)(m0 + lane16) * D_;
  v8f acc[4] = {};
  for (int kc = 0; kc < D_; kc += 128) {
    __syncthreads();
    stage_b(WT, D_, n0, kc, Bs);
    async_wait0();
    __syncthreads();
#pragma unroll
    for (int kkl = 0; kkl < 128; kkl += 32) {
      v16h a = frag_a_f16(arow, kc + kkl, khalf);
#pragma unroll
      for (int nt = 0; nt < 4; ++nt)
        acc[nt] =
            wmma16(a, *(const v16h*)&Bs[nt * 16 + lane16][kkl + khalf * 16],
                   acc[nt]);
    }
  }
  const int rb = (lane & 16) ? 8 : 0;
#pragma unroll
  for (int i = 0; i < 8; ++i)
#pragma unroll
    for (int nt = 0; nt < 4; ++nt) {
      int n = n0 + nt * 16 + lane16;
      Out[(size_t)(m0 + rb + i) * D_ + n] = acc[nt][i] + bias[n];
    }
}

// ---------------------------------------------------------------- launcher
extern "C" void kernel_launch(void* const* d_in, const int* in_sizes, int n_in,
                              void* d_out, int out_size, void* d_ws,
                              size_t ws_size, hipStream_t stream) {
  const float* x     = (const float*)d_in[0];
  const float* Wq    = (const float*)d_in[1];
  const float* Wk    = (const float*)d_in[2];
  const float* Wv    = (const float*)d_in[3];
  const float* convw = (const float*)d_in[4];
  const float* convb = (const float*)d_in[5];
  const float* combw = (const float*)d_in[6];
  const float* combb = (const float*)d_in[7];
  const int*   sidx  = (const int*)d_in[8];

  char* ws = (char*)d_ws;
  size_t off = 0;
  auto alloc = [&](size_t bytes) -> void* {
    void* p = ws + off;
    off += (bytes + 255) & ~(size_t)255;
    return p;
  };
  const size_t ML = (size_t)B_ * L_;  // 32768
  _Float16* Xh     = (_Float16*)alloc(ML * D_ * 2);
  _Float16* Qh     = (_Float16*)alloc(ML * E_ * 2);
  _Float16* Kh     = (_Float16*)alloc(ML * E_ * 2);
  _Float16* Vh     = (_Float16*)alloc(ML * E_ * 2);
  _Float16* Wqt    = (_Float16*)alloc((size_t)D_ * E_ * 2);
  _Float16* Wkt    = (_Float16*)alloc((size_t)D_ * E_ * 2);
  _Float16* Wvt    = (_Float16*)alloc((size_t)D_ * E_ * 2);
  _Float16* convWt = (_Float16*)alloc((size_t)3 * E_ * E_ * 2);
  _Float16* combWt = (_Float16*)alloc((size_t)E_ * D_ * 2);
  _Float16* Ksg    = (_Float16*)alloc((size_t)B_ * H_ * S_ * DK_ * 2);
  _Float16* Vsht   = (_Float16*)alloc((size_t)B_ * H_ * DK_ * S_ * 2);
  int*      idx2   = (int*)alloc(S_ * 4);
  // lifetime-based aliasing (Kh/Vh dead after gather, Qh dead after attn)
  _Float16* Oh   = Kh;  // attention output [B,L,E]
  _Float16* Yc   = Vh;  // conv+elu output  [B,L,E]
  _Float16* Pool = Qh;  // pooled           [B,L/2,E]

  // prep: x->f16, weight transposes, index fusion
  k_xh<<<65536, 256, 0, stream>>>(x, Xh, (int)(ML * D_));
  k_wt<<<1024, 256, 0, stream>>>(Wq, Wqt, D_, E_);
  k_wt<<<1024, 256, 0, stream>>>(Wk, Wkt, D_, E_);
  k_wt<<<1024, 256, 0, stream>>>(Wv, Wvt, D_, E_);
  k_wt<<<3072, 256, 0, stream>>>(convw, convWt, 3 * E_, E_);
  k_wt<<<1024, 256, 0, stream>>>(combw, combWt, E_, D_);
  k_idx2<<<2, 256, 0, stream>>>(sidx, idx2);

  // QKV projections (B tiles async-staged through LDS)
  dim3 gProj(256, 8);  // M/128, N/64
  k_gemm_hw<<<gProj, 256, 0, stream>>>(Xh, Wqt, Qh, D_, E_);
  k_gemm_hw<<<gProj, 256, 0, stream>>>(Xh, Wkt, Kh, D_, E_);
  k_gemm_hw<<<gProj, 256, 0, stream>>>(Xh, Wvt, Vh, D_, E_);

  // sampled K/V tables (score-gather folded into Ksg)
  k_gather<<<8192, 256, 0, stream>>>(Kh, Vh, sidx, idx2, Ksg, Vsht);

  // fused attention
  k_attn<<<2048, 256, 0, stream>>>(Qh, Ksg, Vsht, Oh);

  // conv(k=3) + bias + ELU as GEMM
  k_conv<<<gProj, 256, 0, stream>>>(Oh, convWt, convb, Yc);

  // maxpool by 2
  k_pool<<<32768, 256, 0, stream>>>(Yc, Pool);

  // final dense + bias -> f32 out
  dim3 gFin(128, 8);
  k_final<<<gFin, 256, 0, stream>>>(Pool, combWt, combb, (float*)d_out);
}
